// TransformerBlock_65592740544734
// MI455X (gfx1250) — matrix operand order repaired and checked by reading, not compiled, not run on hardware
//
#include <hip/hip_runtime.h>

// ---------------------------------------------------------------------------
// Point Transformer block, MI455X (gfx1250, wave32, WMMA).
//
// gather(f) @ W == gather(f @ W): K/V computed per-point (32768 rows) then
// gathered -> 16x FLOP cut on those GEMMs.  Remaining per-pair GEMMs map one
// point's M=16 neighbors onto the WMMA M dimension:
//   3 GEMMs/point x (8 N-tiles x 4 K-steps) of v_wmma_f32_16x16x32_bf16.
// ~57 GFLOP vs ~0.6 GB -> compute bound -> bf16 matrix cores (16x16x32 = 8x
// the K-depth per issue vs the f32 16x16x4 WMMA).
//
// Round-2 fixes:
//  * LDS tiles padded to 136 (272B = 68 dwords) so A-fragment ds_load_b128
//    row accesses rotate across the 64 banks (was 16-way conflict at 256B).
//  * B-fragment global loads software-pipelined through a depth-4 register
//    ring: 8 b128 loads in flight instead of wait-per-WMMA.
// ---------------------------------------------------------------------------

typedef __bf16 bf16;
typedef __attribute__((ext_vector_type(8)))  __bf16 v8bf;
typedef __attribute__((ext_vector_type(16))) __bf16 v16bf;
typedef __attribute__((ext_vector_type(8)))  float  v8f;

#define ROWS  32768   // B*N
#define NPTS  16384   // N (per batch)
#define DM    128
#define LDP   136     // padded leading dim for 128-wide bf16 LDS tiles
#define LDF   72      // padded leading dim for 64-wide bf16 LDS tiles

__device__ __forceinline__ v8f vzero8() {
  v8f z;
#pragma unroll
  for (int i = 0; i < 8; ++i) z[i] = 0.0f;
  return z;
}

// A fragment: 16 x K tile, row-major, leading dim lda (LDS, padded).
// lane<16: row=lane, elems 0..7 = K[koff..+7], elems 8..15 = K[koff+16..+23]
// lane>=16: same rows, K offsets +8 / +24   (ISA 7.12.2, 16-bit A 16x32)
__device__ __forceinline__ v16bf load_a_frag(const bf16* A, int lda, int lane, int koff) {
  const int r  = lane & 15;
  const int ko = koff + ((lane >> 4) << 3);
  const bf16* row = A + r * lda + ko;
  v8bf lo = *(const v8bf*)(row);
  v8bf hi = *(const v8bf*)(row + 16);
  v16bf a;
#pragma unroll
  for (int i = 0; i < 8; ++i) { a[i] = lo[i]; a[i + 8] = hi[i]; }
  return a;
}

// B fragment: K x N with weights stored transposed WT[n][k] (dense, ld=ldb).
// lane<16: n=lane, elems 0..7 = K[koff..+7], 8..15 = K[koff+16..+23]; lane>=16: +8 / +24.
__device__ __forceinline__ v16bf load_b_frag(const bf16* Wt, int ldb, int lane,
                                             int nt, int koff) {
  const int n  = nt * 16 + (lane & 15);
  const int ko = koff + ((lane >> 4) << 3);
  const bf16* row = Wt + (size_t)n * ldb + ko;
  v8bf lo = *(const v8bf*)(row);
  v8bf hi = *(const v8bf*)(row + 16);
  v16bf b;
#pragma unroll
  for (int i = 0; i < 8; ++i) { b[i] = lo[i]; b[i + 8] = hi[i]; }
  return b;
}

// C = A(16xKX, LDS lda) @ W(KXx128, global dense) into 8 16x16 C-fragments.
// B fragments stream through a depth-4 register ring (8 b128 loads in flight).
template <int KX>
__device__ __forceinline__ void gemm_tile(const bf16* A, int lda, const bf16* Wt,
                                          int lane, v8f acc[8]) {
  constexpr int NK  = KX / 32;
  constexpr int TOT = 8 * NK;
  v16bf af[NK];
#pragma unroll
  for (int k = 0; k < NK; ++k) af[k] = load_a_frag(A, lda, lane, k * 32);

  v16bf ring[4];
#pragma unroll
  for (int j = 0; j < 4; ++j)
    ring[j] = load_b_frag(Wt, KX, lane, j / NK, (j % NK) * 32);

#pragma unroll
  for (int i = 0; i < TOT; ++i) {
    const int t = i / NK, k = i % NK;
    v16bf bcur = ring[i & 3];
    if (i + 4 < TOT)
      ring[i & 3] = load_b_frag(Wt, KX, lane, (i + 4) / NK, ((i + 4) % NK) * 32);
    acc[t] = __builtin_amdgcn_wmma_f32_16x16x32_bf16(
        false, af[k], false, bcur, (short)0, acc[t], false, false);
  }
}

// Scatter C fragments (16x128) to a padded row-major bf16 tile, + bias, relu.
// C layout: elem r -> (m = r + 8*(lane>>4), c = 16*t + lane&15).
__device__ __forceinline__ void store_c_bf16(bf16* dst, const v8f acc[8],
                                             const float* bias, bool relu, int lane) {
  const int c0 = lane & 15;
  const int mh = (lane >> 4) << 3;
#pragma unroll
  for (int t = 0; t < 8; ++t) {
    const float bv = bias[t * 16 + c0];
#pragma unroll
    for (int r = 0; r < 8; ++r) {
      float v = acc[t][r] + bv;
      if (relu) v = fmaxf(v, 0.0f);
      dst[(mh + r) * LDP + t * 16 + c0] = (bf16)v;
    }
  }
}

// ---------------------------------------------------------------------------
// Kernel 0: weight prep — transpose fp32 (K,N) -> bf16 WT (N,K), dense.
// ---------------------------------------------------------------------------
__global__ void pt_wprep(const float* __restrict__ src, bf16* __restrict__ dst,
                         int K, int N) {
  int e = blockIdx.x * 256 + threadIdx.x;
  if (e >= K * N) return;
  int n = e / K, k = e - n * K;
  dst[e] = (bf16)src[(size_t)k * N + n];
}

// ---------------------------------------------------------------------------
// Kernel 1: per-point GEMMs.  f = feat@fc1+b; Q=f@wq (f32); K=f@wk, V=f@wv
// (bf16, gathered later); SC = feat@sc+b (f32 residual).  One wave / 16 rows.
// ---------------------------------------------------------------------------
__global__ __launch_bounds__(128) void pt_point_gemms(
    const float* __restrict__ feat,
    const bf16* __restrict__ fc1T, const float* __restrict__ fc1b,
    const bf16* __restrict__ wqT,  const bf16* __restrict__ wkT,
    const bf16* __restrict__ wvT,
    const bf16* __restrict__ scT,  const float* __restrict__ scb,
    float* __restrict__ Q, bf16* __restrict__ Kb, bf16* __restrict__ Vb,
    float* __restrict__ SC) {
  const int lane = threadIdx.x & 31;
  const int wave = threadIdx.x >> 5;
  const int rowbase = (blockIdx.x * 4 + wave) * 16;

  __shared__ __align__(16) bf16 s_feat[4][16 * LDF];
  __shared__ __align__(16) bf16 s_f[4][16 * LDP];
  bf16* featB = s_feat[wave];
  bf16* fB    = s_f[wave];

  // stage 16x64 feature tile as bf16 (padded rows)
#pragma unroll 4
  for (int i = 0; i < 32; ++i) {
    int e = lane + 32 * i;
    int m = e >> 6, c = e & 63;
    featB[m * LDF + c] = (bf16)feat[(size_t)(rowbase + m) * 64 + c];
  }

  const int c0 = lane & 15;
  const int mh = (lane >> 4) << 3;
  v8f acc[8];

  // f = feature @ fc1 + b   (no activation)
#pragma unroll
  for (int t = 0; t < 8; ++t) acc[t] = vzero8();
  gemm_tile<64>(featB, LDF, fc1T, lane, acc);
  store_c_bf16(fB, acc, fc1b, false, lane);

  // Q = f @ wq  (f32 out, no bias)
#pragma unroll
  for (int t = 0; t < 8; ++t) acc[t] = vzero8();
  gemm_tile<128>(fB, LDP, wqT, lane, acc);
#pragma unroll
  for (int t = 0; t < 8; ++t)
#pragma unroll
    for (int r = 0; r < 8; ++r)
      Q[(size_t)(rowbase + mh + r) * DM + t * 16 + c0] = acc[t][r];

  // K = f @ wk  (bf16 out)
#pragma unroll
  for (int t = 0; t < 8; ++t) acc[t] = vzero8();
  gemm_tile<128>(fB, LDP, wkT, lane, acc);
#pragma unroll
  for (int t = 0; t < 8; ++t)
#pragma unroll
    for (int r = 0; r < 8; ++r)
      Kb[(size_t)(rowbase + mh + r) * DM + t * 16 + c0] = (bf16)acc[t][r];

  // V = f @ wv
#pragma unroll
  for (int t = 0; t < 8; ++t) acc[t] = vzero8();
  gemm_tile<128>(fB, LDP, wvT, lane, acc);
#pragma unroll
  for (int t = 0; t < 8; ++t)
#pragma unroll
    for (int r = 0; r < 8; ++r)
      Vb[(size_t)(rowbase + mh + r) * DM + t * 16 + c0] = (bf16)acc[t][r];

  // SC = feature @ sc_w + sc_b  (f32 residual)
#pragma unroll
  for (int t = 0; t < 8; ++t) acc[t] = vzero8();
  gemm_tile<64>(featB, LDF, scT, lane, acc);
#pragma unroll
  for (int t = 0; t < 8; ++t) {
    float bv = scb[t * 16 + c0];
#pragma unroll
    for (int r = 0; r < 8; ++r)
      SC[(size_t)(rowbase + mh + r) * DM + t * 16 + c0] = acc[t][r] + bv;
  }
}

// ---------------------------------------------------------------------------
// Kernel 2: fused vector attention.  One wave per 16-point tile; per point:
//   pos  = relu(raw@dW1+b1)@dW2+b2            (WMMA)
//   s    = q - gather(K) + pos
//   h    = relu(s@gW1+gb1)                    (WMMA)
//   sc   = h@gW2+gb2, softmax over M=16       (WMMA + shfl_xor(16))
//   orow = sum_m attn*(gather(V)+pos)
// then per tile:  out = orow@fc2+b + SC       (WMMA)
// ---------------------------------------------------------------------------
__global__ __launch_bounds__(128) void pt_attention(
    const float* __restrict__ xyz, const int* __restrict__ nidx,
    const float* __restrict__ Q, const bf16* __restrict__ Kb,
    const bf16* __restrict__ Vb, const float* __restrict__ SC,
    const float* __restrict__ dW1, const float* __restrict__ dB1,
    const bf16* __restrict__ d2T, const float* __restrict__ dB2,
    const bf16* __restrict__ g1T, const float* __restrict__ gB1,
    const bf16* __restrict__ g2T, const float* __restrict__ gB2,
    const bf16* __restrict__ fc2T, const float* __restrict__ fB2,
    float* __restrict__ out) {
  const int lane = threadIdx.x & 31;
  const int wave = threadIdx.x >> 5;
  const int rowbase = (blockIdx.x * 4 + wave) * 16;
  const int bbase = rowbase & ~(NPTS - 1);   // batch base row

  __shared__ __align__(16) bf16  s_a[4][16 * LDP];  // h1 / s / gathered V
  __shared__ __align__(16) bf16  s_k[4][16 * LDP];  // gathered K / h
  __shared__ __align__(16) bf16  s_p[4][16 * LDP];  // pos
  __shared__ __align__(16) bf16  s_o[4][16 * LDP];  // out rows for fc2
  __shared__ __align__(16) float s_q[4][DM];
  __shared__ __align__(16) float s_raw[4][16 * 4];
  __shared__ int                 s_id[4][16];

  bf16*  aB = s_a[wave];
  bf16*  kB = s_k[wave];
  bf16*  pB = s_p[wave];
  bf16*  oB = s_o[wave];
  float* qB = s_q[wave];
  float* rB = s_raw[wave];
  int*   iB = s_id[wave];

  const int c0 = lane & 15;
  const int mh = (lane >> 4) << 3;
  const float scale = 0.08838834764831845f;  // 1/sqrt(128)
  v8f acc[8];

  for (int p = 0; p < 16; ++p) {
    const int n = rowbase + p;

    // neighbor indices + relative positions (lanes 0..15, one neighbor each)
    if (lane < 16) {
      const int m  = lane;
      const int id = nidx[(size_t)n * 16 + m];
      iB[m] = id;
      const int g = bbase + id;
      __builtin_prefetch(Kb + (size_t)g * DM, 0, 0);   // global_prefetch
      __builtin_prefetch(Vb + (size_t)g * DM, 0, 0);
      const float cx = xyz[(size_t)n * 3 + 0];
      const float cy = xyz[(size_t)n * 3 + 1];
      const float cz = xyz[(size_t)n * 3 + 2];
      rB[m * 4 + 0] = cx - xyz[(size_t)g * 3 + 0];
      rB[m * 4 + 1] = cy - xyz[(size_t)g * 3 + 1];
      rB[m * 4 + 2] = cz - xyz[(size_t)g * 3 + 2];
    }
    // q row (128 f32, 16B per lane)
    ((float4*)qB)[lane] = ((const float4*)(Q + (size_t)n * DM))[lane];

    // h1 = relu(raw @ dW1 + b1)  (K=3, VALU) -> aB
#pragma unroll 4
    for (int i = 0; i < 64; ++i) {
      const int e = lane + 32 * i;
      const int m = e >> 7, c = e & 127;
      float x = dB1[c] + rB[m * 4 + 0] * dW1[c]
                       + rB[m * 4 + 1] * dW1[DM + c]
                       + rB[m * 4 + 2] * dW1[2 * DM + c];
      aB[m * LDP + c] = (bf16)fmaxf(x, 0.0f);
    }

    // gather K rows -> kB (2 lanes per row, 128B each)
    {
      const int m = lane >> 1, half = lane & 1;
      const uint4* src = (const uint4*)(Kb + (size_t)(bbase + iB[m]) * DM + half * 64);
      uint4* dst = (uint4*)(kB + m * LDP + half * 64);
#pragma unroll
      for (int i = 0; i < 8; ++i) dst[i] = src[i];
    }

    // pos = h1 @ dW2 + b2 -> pB
#pragma unroll
    for (int t = 0; t < 8; ++t) acc[t] = vzero8();
    gemm_tile<128>(aB, LDP, d2T, lane, acc);
    store_c_bf16(pB, acc, dB2, false, lane);

    // s = q - k_gathered + pos -> aB
#pragma unroll 4
    for (int i = 0; i < 64; ++i) {
      const int e = lane + 32 * i;
      const int m = e >> 7, c = e & 127;
      float v = qB[c] - (float)kB[m * LDP + c] + (float)pB[m * LDP + c];
      aB[m * LDP + c] = (bf16)v;
    }

    // h = relu(s @ gW1 + gb1) -> kB
#pragma unroll
    for (int t = 0; t < 8; ++t) acc[t] = vzero8();
    gemm_tile<128>(aB, LDP, g1T, lane, acc);
    store_c_bf16(kB, acc, gB1, true, lane);

    // scores = h @ gW2 + gb2 (kept in C fragments)
#pragma unroll
    for (int t = 0; t < 8; ++t) acc[t] = vzero8();
    gemm_tile<128>(kB, LDP, g2T, lane, acc);

    // softmax over M=16 per channel: lane holds 8 rows, partner lane^16 rest
#pragma unroll
    for (int t = 0; t < 8; ++t) {
      const float bv = gB2[t * 16 + c0];
      float mx = -3.4e38f;
#pragma unroll
      for (int r = 0; r < 8; ++r) {
        acc[t][r] = (acc[t][r] + bv) * scale;
        mx = fmaxf(mx, acc[t][r]);
      }
      mx = fmaxf(mx, __shfl_xor(mx, 16, 32));
      float sum = 0.0f;
#pragma unroll
      for (int r = 0; r < 8; ++r) {
        float ev = __expf(acc[t][r] - mx);
        acc[t][r] = ev;
        sum += ev;
      }
      sum += __shfl_xor(sum, 16, 32);
      const float inv = 1.0f / sum;
#pragma unroll
      for (int r = 0; r < 8; ++r) acc[t][r] *= inv;
    }

    // gather V rows -> aB (s no longer needed)
    {
      const int m = lane >> 1, half = lane & 1;
      const uint4* src = (const uint4*)(Vb + (size_t)(bbase + iB[m]) * DM + half * 64);
      uint4* dst = (uint4*)(aB + m * LDP + half * 64);
#pragma unroll
      for (int i = 0; i < 8; ++i) dst[i] = src[i];
    }

    // orow[c] = sum_m attn[m][c] * (V[m][c] + pos[m][c]) -> oB row p
#pragma unroll
    for (int t = 0; t < 8; ++t) {
      float o = 0.0f;
#pragma unroll
      for (int r = 0; r < 8; ++r) {
        const int m = mh + r;
        const int c = t * 16 + c0;
        o += acc[t][r] * ((float)aB[m * LDP + c] + (float)pB[m * LDP + c]);
      }
      o += __shfl_xor(o, 16, 32);
      if (lane < 16) oB[p * LDP + t * 16 + c0] = (bf16)o;
    }
  }

  // out = orow @ fc2 + b + SC  (16 rows batched -> full WMMA tile)
#pragma unroll
  for (int t = 0; t < 8; ++t) acc[t] = vzero8();
  gemm_tile<128>(oB, LDP, fc2T, lane, acc);
#pragma unroll
  for (int t = 0; t < 8; ++t) {
    const float bv = fB2[t * 16 + c0];
#pragma unroll
    for (int r = 0; r < 8; ++r) {
      const size_t off = (size_t)(rowbase + mh + r) * DM + t * 16 + c0;
      out[off] = acc[t][r] + bv + SC[off];
    }
  }
}

// ---------------------------------------------------------------------------
// Launch
// ---------------------------------------------------------------------------
extern "C" void kernel_launch(void* const* d_in, const int* in_sizes, int n_in,
                              void* d_out, int out_size, void* d_ws, size_t ws_size,
                              hipStream_t stream) {
  const float* xyz   = (const float*)d_in[0];
  const float* feat  = (const float*)d_in[1];
  const int*   nidx  = (const int*)d_in[2];
  const float* dW1   = (const float*)d_in[3];
  const float* dB1   = (const float*)d_in[4];
  const float* dW2   = (const float*)d_in[5];
  const float* dB2   = (const float*)d_in[6];
  const float* fc1w  = (const float*)d_in[7];
  const float* fc1b  = (const float*)d_in[8];
  const float* wq    = (const float*)d_in[9];
  const float* wk    = (const float*)d_in[10];
  const float* wv    = (const float*)d_in[11];
  const float* gW1   = (const float*)d_in[12];
  const float* gB1   = (const float*)d_in[13];
  const float* gW2   = (const float*)d_in[14];
  const float* gB2   = (const float*)d_in[15];
  const float* fc2w  = (const float*)d_in[16];
  const float* fc2b  = (const float*)d_in[17];
  const float* scw   = (const float*)d_in[18];
  const float* scb   = (const float*)d_in[19];

  char* ws = (char*)d_ws;
  size_t off = 0;
  float* Q  = (float*)(ws + off); off += (size_t)ROWS * DM * 4;
  bf16*  Kb = (bf16*) (ws + off); off += (size_t)ROWS * DM * 2;
  bf16*  Vb = (bf16*) (ws + off); off += (size_t)ROWS * DM * 2;
  float* SC = (float*)(ws + off); off += (size_t)ROWS * DM * 4;
  bf16* fc1T = (bf16*)(ws + off); off += 128 * 64 * 2;
  bf16* wqT  = (bf16*)(ws + off); off += 128 * 128 * 2;
  bf16* wkT  = (bf16*)(ws + off); off += 128 * 128 * 2;
  bf16* wvT  = (bf16*)(ws + off); off += 128 * 128 * 2;
  bf16* scT  = (bf16*)(ws + off); off += 128 * 64 * 2;
  bf16* d2T  = (bf16*)(ws + off); off += 128 * 128 * 2;
  bf16* g1T  = (bf16*)(ws + off); off += 128 * 128 * 2;
  bf16* g2T  = (bf16*)(ws + off); off += 128 * 128 * 2;
  bf16* fc2T = (bf16*)(ws + off); off += 128 * 128 * 2;

  // Kernel 0: weight transposes to bf16 WT[n][k]
  pt_wprep<<<(64 * 128 + 255) / 256, 256, 0, stream>>>(fc1w, fc1T, 64, 128);
  pt_wprep<<<(128 * 128 + 255) / 256, 256, 0, stream>>>(wq,  wqT, 128, 128);
  pt_wprep<<<(128 * 128 + 255) / 256, 256, 0, stream>>>(wk,  wkT, 128, 128);
  pt_wprep<<<(128 * 128 + 255) / 256, 256, 0, stream>>>(wv,  wvT, 128, 128);
  pt_wprep<<<(64 * 128 + 255) / 256, 256, 0, stream>>>(scw, scT, 64, 128);
  pt_wprep<<<(128 * 128 + 255) / 256, 256, 0, stream>>>(dW2, d2T, 128, 128);
  pt_wprep<<<(128 * 128 + 255) / 256, 256, 0, stream>>>(gW1, g1T, 128, 128);
  pt_wprep<<<(128 * 128 + 255) / 256, 256, 0, stream>>>(gW2, g2T, 128, 128);
  pt_wprep<<<(128 * 128 + 255) / 256, 256, 0, stream>>>(fc2w, fc2T, 128, 128);

  // Kernel 1: per-point GEMMs (2048 wave-tiles, 4 waves/WG)
  pt_point_gemms<<<512, 128, 0, stream>>>(feat, fc1T, fc1b, wqT, wkT, wvT,
                                          scT, scb, Q, Kb, Vb, SC);

  // Kernel 2: fused vector attention
  pt_attention<<<512, 128, 0, stream>>>(xyz, nidx, Q, Kb, Vb, SC,
                                        dW1, dB1, d2T, dB2, g1T, gB1,
                                        g2T, gB2, fc2T, fc2b, (float*)d_out);
}
